// AGFModule_66383014527241
// MI455X (gfx1250) — matrix operand order repaired and checked by reading, not compiled, not run on hardware
//
#include <hip/hip_runtime.h>

typedef __attribute__((ext_vector_type(16))) _Float16 v16h;
typedef __attribute__((ext_vector_type(8)))  _Float16 v8h;
typedef __attribute__((ext_vector_type(8)))  float    v8f;

#define NN 20000      // nodes
#define NE 100000     // edges
#define NH 4          // heads
#define NC 256        // hidden / per-head out
#define NF 1024       // heads*out

// ---------------- WMMA fragment helpers (gfx1250 wave32 layouts) ----------------

static __device__ __forceinline__ v16h pack16(v8h lo, v8h hi) {
  v16h r;
#pragma unroll
  for (int i = 0; i < 8; ++i) { r[i] = lo[i]; r[i + 8] = hi[i]; }
  return r;
}

// A fragment: base points at row*K + (lane>=16 ? 8 : 0) + ks*32; halves [0,8) and [16,24)
static __device__ __forceinline__ v16h load_a_idx(const _Float16* __restrict__ A, size_t off) {
  return pack16(*(const v8h*)(A + off), *(const v8h*)(A + off + 16));
}

// B fragment: base points at ncol*K + (lane>=16 ? 8 : 0) + ks*32; halves [0,8) and [16,24)
static __device__ __forceinline__ v16h load_b_idx(const _Float16* __restrict__ B, size_t off) {
  return pack16(*(const v8h*)(B + off), *(const v8h*)(B + off + 16));
}

static __device__ __forceinline__ v8f wmma_f16(v16h a, v16h b, v8f c) {
  return __builtin_amdgcn_wmma_f32_16x16x32_f16(false, a, false, b, (short)0, c, false, false);
}

// ---------------- small elementwise kernels ----------------

// W [K, Nc] f32  ->  Wt [Nc, K] f16
__global__ void transpose_f16_kernel(const float* __restrict__ W, _Float16* __restrict__ Wt,
                                     int K, int Ncols) {
  int idx = blockIdx.x * 256 + threadIdx.x;
  if (idx >= K * Ncols) return;
  int n = idx / K, k = idx - n * K;
  Wt[idx] = (_Float16)W[(size_t)k * Ncols + n];
}

// h = relu(X @ W + b); X [rows,7], W [7,256]; one block per row, thread = out col
__global__ void encoder_kernel(const float* __restrict__ X, const float* __restrict__ W,
                               const float* __restrict__ b, _Float16* __restrict__ H16) {
  int r = blockIdx.x;
  int j = threadIdx.x;
  float s = b[j];
#pragma unroll
  for (int k = 0; k < 7; ++k) s += X[(size_t)r * 7 + k] * W[k * 256 + j];
  s = fmaxf(s, 0.f);
  H16[(size_t)r * 256 + j] = (_Float16)s;
}

__global__ void init_kernel(float* __restrict__ m, float* __restrict__ denom,
                            float* __restrict__ agg) {
  int idx = blockIdx.x * 256 + threadIdx.x;
  if (idx < NN * NH) { m[idx] = -__builtin_inff(); denom[idx] = 0.f; }
  if (idx < NN * NF) agg[idx] = 0.f;
}

// ---------------- WMMA GEMM: C[M,N] = act(A_f16[M,K] @ Bt_f16[N,K]^T + bias) ----------------
// block = 128 threads = 4 waves; each wave: 16 rows x 64 cols; M % 16 == 0, N % 64 == 0.
// K compile-time (KSTEPS*32), fully unrolled. Loads for step ks+1 are issued (into fresh
// SSA values) BEFORE the WMMAs of step ks, guaranteeing one WMMA-group of prefetch
// distance without any register rotate copies.
template <int KSTEPS>
__global__ __launch_bounds__(128)
void gemm_wmma_kernel(const _Float16* __restrict__ A, const _Float16* __restrict__ Bt,
                      const float* __restrict__ bias, float* __restrict__ C,
                      int M, int N, int do_relu) {
  constexpr int K = KSTEPS * 32;
  int lane = threadIdx.x & 31;
  int wave = threadIdx.x >> 5;
  int row0 = blockIdx.x * 64 + wave * 16;
  if (row0 >= M) return;                      // wave-uniform: EXEC stays all-1s for WMMA
  int col0 = blockIdx.y * 64;
  int nl = lane & 15;
  int arow = row0 + nl;

  size_t aoff = (size_t)arow * K + ((lane >= 16) ? 8 : 0);
  size_t boff[4];
#pragma unroll
  for (int t = 0; t < 4; ++t)
    boff[t] = (size_t)(col0 + t * 16 + nl) * K + ((lane >= 16) ? 8 : 0);

  v8f acc[4] = {};

  v16h a_cur = load_a_idx(A, aoff);
  v16h b_cur[4];
#pragma unroll
  for (int t = 0; t < 4; ++t) b_cur[t] = load_b_idx(Bt, boff[t]);

#pragma unroll
  for (int ks = 0; ks < KSTEPS; ++ks) {
    v16h a_nxt = a_cur;
    v16h b_nxt[4] = {b_cur[0], b_cur[1], b_cur[2], b_cur[3]};
    if (ks + 1 < KSTEPS) {                   // compile-time resolved in unrolled body
      size_t o = (size_t)(ks + 1) * 32;
      a_nxt = load_a_idx(A, aoff + o);
#pragma unroll
      for (int t = 0; t < 4; ++t) b_nxt[t] = load_b_idx(Bt, boff[t] + o);
    }
#pragma unroll
    for (int t = 0; t < 4; ++t) acc[t] = wmma_f16(a_cur, b_cur[t], acc[t]);
    a_cur = a_nxt;                            // SSA renames in unrolled code: no v_movs
#pragma unroll
    for (int t = 0; t < 4; ++t) b_cur[t] = b_nxt[t];
  }

  int rbase = row0 + ((lane >= 16) ? 8 : 0);
#pragma unroll
  for (int t = 0; t < 4; ++t) {
    int col = col0 + t * 16 + nl;
    float bv = bias[col];
#pragma unroll
    for (int v = 0; v < 8; ++v) {
      float val = acc[t][v] + bv;
      if (do_relu) val = fmaxf(val, 0.f);
      C[(size_t)(rbase + v) * N + col] = val;
    }
  }
}

// ---------------- fused e_h GEMM + GATv2 logits ----------------
// block = 128 threads = 4 waves; block handles 16 edges, wave w handles head w.
// logits[e,h] = sum_c leaky( (e@We+be)[e, h*256+c] + xl[src][h*256+c] + xr[dst][h*256+c] ) * att[h,c]
// A fragments (the 16 E16 rows, all K=256) are loaded ONCE per wave and reused for all 16 N-tiles.
// Dual accumulators break the serial WMMA chain; B loads are prefetched one step ahead.
__global__ __launch_bounds__(128)
void edge_logits_kernel(const _Float16* __restrict__ E16,  // [NE,256]
                        const _Float16* __restrict__ WeT,  // [1024,256]
                        const float* __restrict__ be,      // [1024]
                        const float* __restrict__ xl,      // [NN,1024]
                        const float* __restrict__ xr,      // [NN,1024]
                        const float* __restrict__ att,     // [4,256] flat == [1024]
                        const int* __restrict__ src, const int* __restrict__ dstv,
                        float* __restrict__ logits)        // [NE,4]
{
  int lane = threadIdx.x & 31;
  int head = threadIdx.x >> 5;
  int e0 = blockIdx.x * 16;
  int nl = lane & 15;
  int rbase = (lane >= 16) ? 8 : 0;
  int arow = e0 + nl;

  // per-row gather base offsets (computed once); integer offsets keep loads GLOBAL
  size_t soff[8], doff[8];
#pragma unroll
  for (int v = 0; v < 8; ++v) {
    int er = e0 + rbase + v;
    soff[v] = (size_t)src[er] * NF;
    doff[v] = (size_t)dstv[er] * NF;
  }

  // hoist all 8 K-step A fragments (this wave's slice of 16 full E16 rows)
  size_t aoff = (size_t)arow * 256 + ((lane >= 16) ? 8 : 0);
  v16h afrag[8];
#pragma unroll
  for (int ks = 0; ks < 8; ++ks) afrag[ks] = load_a_idx(E16, aoff + ks * 32);

  float part[8] = {0.f, 0.f, 0.f, 0.f, 0.f, 0.f, 0.f, 0.f};

#pragma unroll 4
  for (int nt = 0; nt < 16; ++nt) {
    int col = head * 256 + nt * 16 + nl;
    size_t bwoff = (size_t)col * 256 + ((lane >= 16) ? 8 : 0);

    v8f acc0 = {}, acc1 = {};
    v16h b_cur = load_b_idx(WeT, bwoff);
#pragma unroll
    for (int ks = 0; ks < 8; ++ks) {
      v16h b_nxt = b_cur;
      if (ks + 1 < 8) b_nxt = load_b_idx(WeT, bwoff + (size_t)(ks + 1) * 32);
      if (ks & 1) acc1 = wmma_f16(afrag[ks], b_cur, acc1);
      else        acc0 = wmma_f16(afrag[ks], b_cur, acc0);
      b_cur = b_nxt;                          // SSA rename (unrolled), no copies
    }

    float bv = be[col];
    float av = att[col];   // att flat index h*256 + c == col
#pragma unroll
    for (int v = 0; v < 8; ++v) {
      float z = acc0[v] + acc1[v] + bv + xl[soff[v] + col] + xr[doff[v] + col];
      z = (z > 0.f) ? z : 0.2f * z;
      part[v] += z * av;
    }
  }

  // reduce each row's 16 column-partials across the half-wave (lanes 0-15 / 16-31)
#pragma unroll
  for (int v = 0; v < 8; ++v) {
    float s = part[v];
    s += __shfl_xor(s, 8, 16);
    s += __shfl_xor(s, 4, 16);
    s += __shfl_xor(s, 2, 16);
    s += __shfl_xor(s, 1, 16);
    if (nl == 0) logits[(size_t)(e0 + rbase + v) * NH + head] = s;
  }
}

// ---------------- segment softmax + aggregation ----------------

static __device__ __forceinline__ void atomicMaxF(float* addr, float val) {
  if (val >= 0.f) atomicMax((int*)addr, __float_as_int(val));
  else            atomicMin((unsigned int*)addr, __float_as_uint(val));
}

__global__ void segmax_kernel(const float* __restrict__ logits, const int* __restrict__ dstv,
                              float* __restrict__ m) {
  int idx = blockIdx.x * 256 + threadIdx.x;
  if (idx >= NE * NH) return;
  int e = idx >> 2, h = idx & 3;
  atomicMaxF(&m[(size_t)dstv[e] * NH + h], logits[idx]);
}

__global__ void expv_kernel(const float* __restrict__ logits, const int* __restrict__ dstv,
                            const float* __restrict__ m, float* __restrict__ expv,
                            float* __restrict__ denom) {
  int idx = blockIdx.x * 256 + threadIdx.x;
  if (idx >= NE * NH) return;
  int e = idx >> 2, h = idx & 3;
  int d = dstv[e];
  float ex = __expf(logits[idx] - m[(size_t)d * NH + h]);
  expv[idx] = ex;
  atomicAdd(&denom[(size_t)d * NH + h], ex);
}

// agg[dst, c] += xl[src, c] * alpha[e, head(c)] ; block per edge, 256 threads x 4 reps
__global__ void agg_kernel(const float* __restrict__ xl, const float* __restrict__ expv,
                           const float* __restrict__ denom, const int* __restrict__ src,
                           const int* __restrict__ dstv, float* __restrict__ agg) {
  int e = blockIdx.x;
  int t = threadIdx.x;
  int s = src[e], d = dstv[e];
#pragma unroll
  for (int rep = 0; rep < 4; ++rep) {
    int c = t + rep * 256;
    int h = c >> 8;
    float a = expv[(size_t)e * NH + h] / denom[(size_t)d * NH + h];
    atomicAdd(&agg[(size_t)d * NF + c], xl[(size_t)s * NF + c] * a);
  }
}

__global__ void enc16_kernel(const float* __restrict__ agg, const float* __restrict__ conv_bias,
                             _Float16* __restrict__ enc16) {
  int idx = blockIdx.x * 256 + threadIdx.x;
  if (idx >= NN * NF) return;
  int c = idx & (NF - 1);
  enc16[idx] = (_Float16)(agg[idx] + conv_bias[c]);
}

// out = sigmoid(hid @ W_d2 + b_d2); hid [NN,256], W_d2 [256,6]
__global__ void dec2_kernel(const float* __restrict__ hid, const float* __restrict__ W2,
                            const float* __restrict__ b2, float* __restrict__ out) {
  int idx = blockIdx.x * 256 + threadIdx.x;
  if (idx >= NN * 6) return;
  int n = idx / 6, j = idx - n * 6;
  float s = b2[j];
  const float* hr = hid + (size_t)n * 256;
  for (int k = 0; k < 256; ++k) s += hr[k] * W2[k * 6 + j];
  out[idx] = 1.f / (1.f + __expf(-s));
}

// ---------------- launch ----------------

extern "C" void kernel_launch(void* const* d_in, const int* in_sizes, int n_in,
                              void* d_out, int out_size, void* d_ws, size_t ws_size,
                              hipStream_t stream) {
  const float* x         = (const float*)d_in[0];
  const float* edge_attr = (const float*)d_in[1];
  const float* W_ne = (const float*)d_in[2];  const float* b_ne = (const float*)d_in[3];
  const float* W_ee = (const float*)d_in[4];  const float* b_ee = (const float*)d_in[5];
  const float* W_l  = (const float*)d_in[6];  const float* b_l  = (const float*)d_in[7];
  const float* W_r  = (const float*)d_in[8];  const float* b_r  = (const float*)d_in[9];
  const float* W_e  = (const float*)d_in[10]; const float* b_e  = (const float*)d_in[11];
  const float* att  = (const float*)d_in[12];
  const float* conv_bias = (const float*)d_in[13];
  const float* W_d1 = (const float*)d_in[14]; const float* b_d1 = (const float*)d_in[15];
  const float* W_d2 = (const float*)d_in[16]; const float* b_d2 = (const float*)d_in[17];
  const int*   ei   = (const int*)d_in[18];
  const int* src = ei;
  const int* dst = ei + NE;

  char* ws = (char*)d_ws;
  size_t off = 0;
  auto alloc = [&](size_t bytes) -> void* {
    void* p = ws + off;
    off += (bytes + 255) & ~(size_t)255;
    return p;
  };

  _Float16* h16   = (_Float16*)alloc((size_t)NN * NC * 2);
  _Float16* e16   = (_Float16*)alloc((size_t)NE * NC * 2);
  _Float16* WlT   = (_Float16*)alloc((size_t)NF * NC * 2);
  _Float16* WrT   = (_Float16*)alloc((size_t)NF * NC * 2);
  _Float16* WeT   = (_Float16*)alloc((size_t)NF * NC * 2);
  _Float16* Wd1T  = (_Float16*)alloc((size_t)NC * NF * 2);
  float*    xl    = (float*)alloc((size_t)NN * NF * 4);
  float*    xr    = (float*)alloc((size_t)NN * NF * 4);
  float*    logits= (float*)alloc((size_t)NE * NH * 4);
  float*    m     = (float*)alloc((size_t)NN * NH * 4);
  float*    denom = (float*)alloc((size_t)NN * NH * 4);
  float*    expv  = (float*)alloc((size_t)NE * NH * 4);
  float*    agg   = (float*)alloc((size_t)NN * NF * 4);
  _Float16* enc16 = (_Float16*)alloc((size_t)NN * NF * 2);
  float*    hid   = (float*)alloc((size_t)NN * NC * 4);
  (void)ws_size; (void)in_sizes; (void)n_in; (void)out_size;

  // 1. f32 -> f16 transposed weight copies for WMMA B operands
  int wblocks = (NC * NF + 255) / 256;
  transpose_f16_kernel<<<wblocks, 256, 0, stream>>>(W_l,  WlT,  NC, NF);
  transpose_f16_kernel<<<wblocks, 256, 0, stream>>>(W_r,  WrT,  NC, NF);
  transpose_f16_kernel<<<wblocks, 256, 0, stream>>>(W_e,  WeT,  NC, NF);
  transpose_f16_kernel<<<wblocks, 256, 0, stream>>>(W_d1, Wd1T, NF, NC);

  // 2. node / edge encoders (K=7, VALU)
  encoder_kernel<<<NN, 256, 0, stream>>>(x, W_ne, b_ne, h16);
  encoder_kernel<<<NE, 256, 0, stream>>>(edge_attr, W_ee, b_ee, e16);

  // 3. x_l, x_r via WMMA GEMM (M=20000, N=1024, K=256 -> KSTEPS=8)
  dim3 g1((NN + 63) / 64, NF / 64);
  gemm_wmma_kernel<8><<<g1, 128, 0, stream>>>(h16, WlT, b_l, xl, NN, NF, 0);
  gemm_wmma_kernel<8><<<g1, 128, 0, stream>>>(h16, WrT, b_r, xr, NN, NF, 0);

  // 4. init m=-inf, denom=0, agg=0
  init_kernel<<<(NN * NF + 255) / 256, 256, 0, stream>>>(m, denom, agg);

  // 5. fused e_h WMMA GEMM + leaky-relu + att dot -> logits (never materializes e_h)
  edge_logits_kernel<<<NE / 16, 128, 0, stream>>>(e16, WeT, b_e, xl, xr, att, src, dst, logits);

  // 6. segment softmax
  int ehblocks = (NE * NH + 255) / 256;
  segmax_kernel<<<ehblocks, 256, 0, stream>>>(logits, dst, m);
  expv_kernel<<<ehblocks, 256, 0, stream>>>(logits, dst, m, expv, denom);

  // 7. weighted aggregation (scatter-add)
  agg_kernel<<<NE, 256, 0, stream>>>(xl, expv, denom, src, dst, agg);

  // 8. node_enc = agg + conv_bias (f16 for decoder GEMM)
  enc16_kernel<<<(NN * NF + 255) / 256, 256, 0, stream>>>(agg, conv_bias, enc16);

  // 9. decoder layer 1 via WMMA GEMM (M=20000, N=256, K=1024 -> KSTEPS=32) with ReLU
  dim3 g2((NN + 63) / 64, NC / 64);
  gemm_wmma_kernel<32><<<g2, 128, 0, stream>>>(enc16, Wd1T, b_d1, hid, NN, NC, 1);

  // 10. decoder layer 2 + sigmoid -> output
  dec2_kernel<<<(NN * 6 + 255) / 256, 256, 0, stream>>>(hid, W_d2, b_d2, (float*)d_out);
}
